// NewAttention_18511309046122
// MI455X (gfx1250) — matrix operand order repaired and checked by reading, not compiled, not run on hardware
//
#include <hip/hip_runtime.h>
#include <stdint.h>

typedef __bf16 bf16_t;
typedef __attribute__((ext_vector_type(16))) __bf16 v16bf;
typedef __attribute__((ext_vector_type(8)))  __bf16 v8bf;
typedef __attribute__((ext_vector_type(4)))  __bf16 v4bf;
typedef __attribute__((ext_vector_type(8)))  float   v8f;
typedef __attribute__((ext_vector_type(4)))  float   v4f;
typedef unsigned int u32;
typedef __attribute__((ext_vector_type(4))) u32 v4u;
typedef __attribute__((ext_vector_type(8))) int v8i;
typedef __attribute__((ext_vector_type(4))) int v4i;

#define DEV static __device__ __forceinline__

// ---------- helpers ----------
DEV bf16_t f2bf(float f) {
  uint32_t u = __builtin_bit_cast(uint32_t, f);
  uint32_t r = u + 0x7FFFu + ((u >> 16) & 1u);   // round-to-nearest-even
  return __builtin_bit_cast(bf16_t, (uint16_t)(r >> 16));
}

// A fragment (bf16 16x32, MxK) from row-major [*, ld], rows m0.., cols k0..
// lanes 0-15 -> M=lane, K=k0+{0..7},{16..23}; lanes16-31 -> K=k0+{8..15},{24..31}
DEV v16bf load_frag_a(const bf16_t* base, int ld, int m0, int k0, int lane) {
  const int row = m0 + (lane & 15);
  const int kb  = k0 + ((lane & 16) ? 8 : 0);
  const bf16_t* p = base + (size_t)row * ld + kb;
  v8bf lo = *(const v8bf*)(p);
  v8bf hi = *(const v8bf*)(p + 16);
  v16bf r;
#pragma unroll
  for (int i = 0; i < 8; ++i) { r[i] = lo[i]; r[8 + i] = hi[i]; }
  return r;
}

// B fragment (bf16 32x16, KxN): B(k,n) = base[(n0+n)*ld + k0+k]  (row-major [N,K])
// lane -> N=lane&15; lanes0-15 hold K=k0+{0..7},{16..23}; lanes16-31 hold K=k0+{8..15},{24..31}.
DEV v16bf load_frag_b(const bf16_t* base, int ld, int n0, int k0, int lane) {
  const int col = n0 + (lane & 15);
  const int kb  = k0 + ((lane & 16) ? 8 : 0);
  const bf16_t* p = base + (size_t)col * ld + kb;
  v8bf lo = *(const v8bf*)(p);
  v8bf hi = *(const v8bf*)(p + 16);
  v16bf r;
#pragma unroll
  for (int i = 0; i < 8; ++i) { r[i] = lo[i]; r[8 + i] = hi[i]; }
  return r;
}

DEV v8f wmma_bf16(v16bf a, v16bf b, v8f c) {
  return __builtin_amdgcn_wmma_f32_16x16x32_bf16(false, a, false, b, (short)0, c, false, false);
}

DEV u32 lds_off(const void* p) { return (u32)(uintptr_t)p; }  // low 32 bits = LDS byte offset

// Tensor Data Mover: 2D tile load (elements are 2-byte). D# per ISA ch.8.
// tile_d1 == 0 -> 1D tile. Dims/strides in data_size (2B) units.
// This toolchain exposes the 6-arg builtin: (v4u, v8i, v4i, v4i, v8i, i32 cpol).
DEV void tdm_load_2d(u32 lds_addr, const void* gptr,
                     u32 tensor_d0, u32 tensor_d1, u32 stride0,
                     u32 tile_d0, u32 tile_d1) {
  const uint64_t ga = (uint64_t)(uintptr_t)gptr;
  v4u g0;
  g0[0] = 1u;                                        // count=1 (valid), user mode
  g0[1] = lds_addr;                                  // lds_addr [63:32]
  g0[2] = (u32)ga;                                   // global_addr [95:64]
  g0[3] = ((u32)(ga >> 32) & 0x01FFFFFFu) | (2u << 30); // addr[56:32], type=2 (image)
  v8i g1;
  g1[0] = (int)(1u << 16);                           // wg_mask=0, data_size=1 (2B)
  g1[1] = (int)(tensor_d0 << 16);                    // tensor_dim0[15:0] @ bits 63:48
  g1[2] = (int)((tensor_d0 >> 16) | (tensor_d1 << 16)); // dim0 hi / dim1 lo
  g1[3] = (int)((tensor_d1 >> 16) | (tile_d0 << 16));   // dim1 hi / tile_dim0
  g1[4] = (int)(tile_d1 & 0xFFFFu);                  // tile_dim1; tile_dim2=0
  g1[5] = (int)stride0;                              // tensor_dim0_stride lo32
  g1[6] = 0;                                         // stride0 hi16 / stride1 lo16
  g1[7] = 0;                                         // stride1 hi32
  v4i g2; g2[0] = 1; g2[1] = 0; g2[2] = 0; g2[3] = 0; // tensor_dim2=1, rest 0
  v4i g3; g3[0] = 0; g3[1] = 0; g3[2] = 0; g3[3] = 0;
  v8i g4; g4[0] = 0; g4[1] = 0; g4[2] = 0; g4[3] = 0; // zero-filled extra group
  g4[4] = 0; g4[5] = 0; g4[6] = 0; g4[7] = 0;
  __builtin_amdgcn_tensor_load_to_lds(g0, g1, g2, g3, g4, 0);
}

// ---------- kernels ----------
__global__ void __launch_bounds__(256) cvt_f32_bf16(const float* __restrict__ in,
                                                    bf16_t* __restrict__ out, int n) {
  int i = (blockIdx.x * blockDim.x + threadIdx.x) * 4;
  if (i + 3 < n) {
    v4f v = *(const v4f*)(in + i);
    v4bf o;
#pragma unroll
    for (int j = 0; j < 4; ++j) o[j] = f2bf(v[j]);
    *(v4bf*)(out + i) = o;
  }
}

// QKV projection: [8192,512] x [1536,512]^T; 32x64 tile per wave (8 WMMA / 12 b128 loads).
// Epilogue scatters into Q[b,h,s,32] (pre-scaled), K[b,h,s,32], Vt[b,h,32,s].
__global__ void __launch_bounds__(128) qkv_gemm(const bf16_t* __restrict__ X,
                                                const bf16_t* __restrict__ W,
                                                bf16_t* __restrict__ Qb,
                                                bf16_t* __restrict__ Kb,
                                                bf16_t* __restrict__ Vt) {
  const int lane = threadIdx.x & 31;
  const int wv   = threadIdx.x >> 5;
  const int m0   = blockIdx.y * 128 + wv * 32;
  const int n0   = blockIdx.x * 64;
  v8f acc[2][4] = {};
  for (int k0 = 0; k0 < 512; k0 += 32) {
    v16bf a0 = load_frag_a(X, 512, m0, k0, lane);
    v16bf a1 = load_frag_a(X, 512, m0 + 16, k0, lane);
#pragma unroll
    for (int j = 0; j < 4; ++j) {
      v16bf b = load_frag_b(W, 512, n0 + 16 * j, k0, lane);
      acc[0][j] = wmma_bf16(a0, b, acc[0][j]);
      acc[1][j] = wmma_bf16(a1, b, acc[1][j]);
    }
  }
  const int half8 = (lane & 16) ? 8 : 0;
  const float qscale = 0.17677669529663687f;   // 32^-0.5
#pragma unroll
  for (int i = 0; i < 2; ++i) {
#pragma unroll
    for (int j = 0; j < 4; ++j) {
      const int n = n0 + 16 * j + (lane & 15);
      const int sector = n >> 9;               // 0=q 1=k 2=v
      const int nn = n & 511;
      const int h = nn >> 5, dd = nn & 31;
#pragma unroll
      for (int r = 0; r < 8; ++r) {
        const int row = m0 + 16 * i + r + half8;
        const int b_ = row >> 11, s = row & 2047;
        const int bh = b_ * 16 + h;
        const float v = acc[i][j][r];
        if (sector == 0)      Qb[((size_t)bh * 2048 + s) * 32 + dd] = f2bf(v * qscale);
        else if (sector == 1) Kb[((size_t)bh * 2048 + s) * 32 + dd] = f2bf(v);
        else                  Vt[((size_t)bh * 32 + dd) * 2048 + s] = f2bf(v);
      }
    }
  }
}

// Flash attention: one wave owns 16 query rows of one (b,h); d=32 == WMMA K.
// K/V 32x32 bf16 blocks staged into LDS by the Tensor Data Mover, double-buffered.
__global__ void __launch_bounds__(128) attn_kernel(const bf16_t* __restrict__ Qb,
                                                   const bf16_t* __restrict__ Kb,
                                                   const bf16_t* __restrict__ Vt,
                                                   bf16_t* __restrict__ attn) {
  __shared__ __align__(16) bf16_t kbuf[4][2][32 * 32];   // [wave][buf][key][dd]
  __shared__ __align__(16) bf16_t vbuf[4][2][32 * 32];   // [wave][buf][dd][key]
  __shared__ __align__(16) bf16_t pbuf[4][16 * 32];      // P tile bounce
  const int lane = threadIdx.x & 31;
  const int wv   = threadIdx.x >> 5;
  const int t    = blockIdx.x * 4 + wv;
  const int qt   = t & 127;
  const int bh   = t >> 7;
  const bf16_t* Q = Qb + (size_t)bh * 2048 * 32;
  const bf16_t* K = Kb + (size_t)bh * 2048 * 32;
  const bf16_t* V = Vt + (size_t)bh * 32 * 2048;
  bf16_t* pl = pbuf[wv];

  // prologue: TDM-stage block 0 into buffer 0
  // K block: 32 contiguous rows of 32 -> 1D tile of 1024 elements
  tdm_load_2d(lds_off(kbuf[wv][0]), K, 65536u, 1u, 65536u, 1024u, 0u);
  // V block: 32 rows (dd) x 32 keys, row stride 2048 elements
  tdm_load_2d(lds_off(vbuf[wv][0]), V, 2048u, 32u, 2048u, 32u, 32u);

  const v16bf aq = load_frag_a(Q, 32, qt * 16, 0, lane);   // whole d, loaded once
  v8f o0 = {}, o1 = {};
  float mrow[8], lrow[8];
#pragma unroll
  for (int r = 0; r < 8; ++r) { mrow[r] = -3.0e38f; lrow[r] = 0.0f; }

  const int row0 = (lane & 16) ? 8 : 0;
  const int colc = lane & 15;

  for (int i = 0; i < 64; ++i) {
    const int cur = i & 1;
    if (i + 1 < 64) {                       // uniform: stage next block, then wait for current
      const int kb1 = (i + 1) * 32;
      tdm_load_2d(lds_off(kbuf[wv][cur ^ 1]), K + (size_t)kb1 * 32,
                  65536u, 1u, 65536u, 1024u, 0u);
      tdm_load_2d(lds_off(vbuf[wv][cur ^ 1]), V + kb1,
                  2048u, 32u, 2048u, 32u, 32u);
      __builtin_amdgcn_s_wait_tensorcnt(2); // in-order: current buffer's 2 loads are done
    } else {
      __builtin_amdgcn_s_wait_tensorcnt(0);
    }
    const bf16_t* kl = kbuf[wv][cur];
    const bf16_t* vl = vbuf[wv][cur];

    v16bf bk0 = load_frag_b(kl, 32, 0, 0, lane);
    v16bf bk1 = load_frag_b(kl, 32, 16, 0, lane);
    v8f c0 = {}, c1 = {};
    c0 = wmma_bf16(aq, bk0, c0);            // logits (scale folded into Q)
    c1 = wmma_bf16(aq, bk1, c1);

    float p0[8], p1[8];
#pragma unroll
    for (int r = 0; r < 8; ++r) {           // online softmax, row = r + row0
      float mx = fmaxf(c0[r], c1[r]);
      mx = fmaxf(mx, __shfl_xor(mx, 1));
      mx = fmaxf(mx, __shfl_xor(mx, 2));
      mx = fmaxf(mx, __shfl_xor(mx, 4));
      mx = fmaxf(mx, __shfl_xor(mx, 8));
      const float mnew = fmaxf(mrow[r], mx);
      const float corr = __expf(mrow[r] - mnew);
      const float e0 = __expf(c0[r] - mnew);
      const float e1 = __expf(c1[r] - mnew);
      float sum = e0 + e1;
      sum += __shfl_xor(sum, 1);
      sum += __shfl_xor(sum, 2);
      sum += __shfl_xor(sum, 4);
      sum += __shfl_xor(sum, 8);
      lrow[r] = lrow[r] * corr + sum;
      o0[r] *= corr; o1[r] *= corr;
      mrow[r] = mnew;
      p0[r] = e0; p1[r] = e1;
    }
    // C-layout -> A-layout bounce through LDS
#pragma unroll
    for (int r = 0; r < 8; ++r) {
      pl[(row0 + r) * 32 + colc]      = f2bf(p0[r]);
      pl[(row0 + r) * 32 + colc + 16] = f2bf(p1[r]);
    }
    asm volatile("s_wait_dscnt 0" ::: "memory");
    v16bf ap  = load_frag_a(pl, 32, 0, 0, lane);
    v16bf bv0 = load_frag_b(vl, 32, 0, 0, lane);
    v16bf bv1 = load_frag_b(vl, 32, 16, 0, lane);
    o0 = wmma_bf16(ap, bv0, o0);
    o1 = wmma_bf16(ap, bv1, o1);
  }

  const int b_ = bh >> 4, h = bh & 15;
#pragma unroll
  for (int r = 0; r < 8; ++r) {
    const float inv = 1.0f / lrow[r];
    const int s = qt * 16 + r + row0;
    const size_t rowbase = ((size_t)b_ * 2048 + s) * 512 + h * 32;
    attn[rowbase + colc]      = f2bf(o0[r] * inv);
    attn[rowbase + colc + 16] = f2bf(o1[r] * inv);
  }
}

// Output projection: [8192,512] x [512,512]^T -> fp32 d_out; 32x64 tile per wave.
__global__ void __launch_bounds__(128) out_gemm(const bf16_t* __restrict__ A,
                                                const bf16_t* __restrict__ W,
                                                float* __restrict__ out) {
  const int lane = threadIdx.x & 31;
  const int wv   = threadIdx.x >> 5;
  const int m0   = blockIdx.y * 128 + wv * 32;
  const int n0   = blockIdx.x * 64;
  v8f acc[2][4] = {};
  for (int k0 = 0; k0 < 512; k0 += 32) {
    v16bf a0 = load_frag_a(A, 512, m0, k0, lane);
    v16bf a1 = load_frag_a(A, 512, m0 + 16, k0, lane);
#pragma unroll
    for (int j = 0; j < 4; ++j) {
      v16bf b = load_frag_b(W, 512, n0 + 16 * j, k0, lane);
      acc[0][j] = wmma_bf16(a0, b, acc[0][j]);
      acc[1][j] = wmma_bf16(a1, b, acc[1][j]);
    }
  }
  const int half8 = (lane & 16) ? 8 : 0;
#pragma unroll
  for (int i = 0; i < 2; ++i) {
#pragma unroll
    for (int j = 0; j < 4; ++j) {
      const int n = n0 + 16 * j + (lane & 15);
#pragma unroll
      for (int r = 0; r < 8; ++r) {
        const int row = m0 + 16 * i + r + half8;
        out[(size_t)row * 512 + n] = acc[i][j][r];
      }
    }
  }
}

// ---------- launcher ----------
extern "C" void kernel_launch(void* const* d_in, const int* in_sizes, int n_in,
                              void* d_out, int out_size, void* d_ws, size_t ws_size,
                              hipStream_t stream) {
  const float* x  = (const float*)d_in[0];   // [4,2048,512]
  const float* wi = (const float*)d_in[1];   // [1536,512]
  const float* wo = (const float*)d_in[2];   // [512,512]
  float* out = (float*)d_out;

  char* ws = (char*)d_ws;
  bf16_t* xb   = (bf16_t*)(ws + 0);           // 8  MB
  bf16_t* wqkv = (bf16_t*)(ws + 8388608);     // 1.5MB
  bf16_t* wout = (bf16_t*)(ws + 9961472);     // 0.5MB
  bf16_t* Qb   = (bf16_t*)(ws + 10485760);    // 8  MB
  bf16_t* Kb   = (bf16_t*)(ws + 18874368);    // 8  MB
  bf16_t* Vt   = (bf16_t*)(ws + 27262976);    // 8  MB
  bf16_t* attn = (bf16_t*)(ws + 35651584);    // 8  MB

  cvt_f32_bf16<<<(8192 * 512 / 4) / 256, 256, 0, stream>>>(x, xb, 8192 * 512);
  cvt_f32_bf16<<<(1536 * 512 / 4) / 256, 256, 0, stream>>>(wi, wqkv, 1536 * 512);
  cvt_f32_bf16<<<(512 * 512 / 4) / 256, 256, 0, stream>>>(wo, wout, 512 * 512);

  dim3 g1(1536 / 64, 8192 / 128);
  qkv_gemm<<<g1, 128, 0, stream>>>(xb, wqkv, Qb, Kb, Vt);

  attn_kernel<<<(4 * 16 * 128) / 4, 128, 0, stream>>>(Qb, Kb, Vt, attn);

  dim3 g2(512 / 64, 8192 / 128);
  out_gemm<<<g2, 128, 0, stream>>>(attn, wout, out);
}
